// DepthSample_46591805227678
// MI455X (gfx1250) — matrix operand order repaired and checked by reading, not compiled, not run on hardware
//
#include <hip/hip_runtime.h>
#include <hip/hip_bf16.h>
#include <math.h>

#define B_  8
#define C_  128
#define H_  192
#define W_  192
#define NH_ 16
#define HW_ (H_*W_)
#define K_  (9*C_)          // 1152
#define KSTEPS (K_/32)      // 36
#define NTILES_X (W_/16)    // 12
#define NPIXTILES (B_*H_*NTILES_X) // 18432

typedef __attribute__((ext_vector_type(16))) __bf16 v16bf;
typedef __attribute__((ext_vector_type(8)))  float  v8f;

union AFrag { v16bf v; __bf16 e[16]; unsigned short s[16]; uint4 q[2]; };

__device__ __forceinline__ unsigned short f2bf(float f) {
    union { float f; unsigned u; } cv; cv.f = f;
    unsigned u = cv.u;
    unsigned r = u + 0x7FFFu + ((u >> 16) & 1u);   // round-to-nearest-even
    return (unsigned short)(r >> 16);
}

// ---------------------------------------------------------------------------
// Kernel 1: pre-swizzle weights into bf16 WMMA A-fragments.
// A-fragment layout used here (16-bit A 16x32, wave32; contiguous k, not AMD's table order):
//   M = mtile*16 + (lane&15),  K = 32*t + 16*(lane>>4) + j
// Stored as aFrag[((mtile*KSTEPS + t)*32 + lane)*16 + j]
// K ordering: K = tap*128 + c,  tap = ky*3+kx.
// ---------------------------------------------------------------------------
__global__ void __launch_bounds__(256)
prep_weights_kernel(const float* __restrict__ wOff,   // [27][128][3][3]
                    const float* __restrict__ wDcn,   // [128][128][3][3]
                    unsigned short* __restrict__ aOff, // 2*KSTEPS*512
                    unsigned short* __restrict__ aDcn) // 8*KSTEPS*512
{
    int idx = blockIdx.x * 256 + threadIdx.x;
    const int total_dcn = 8 * KSTEPS * 512;
    const int total_off = 2 * KSTEPS * 512;
    if (idx < total_dcn) {
        int m    = idx / (KSTEPS * 512);
        int r    = idx % (KSTEPS * 512);
        int t    = r / 512;
        int lane = (r % 512) / 16;
        int j    = r % 16;
        int M  = m * 16 + (lane & 15);
        int Kk = 32 * t + 16 * (lane >> 4) + j;
        int tap = Kk >> 7, c = Kk & 127;
        int ky = tap / 3, kx = tap - 3 * (tap / 3);
        aDcn[idx] = f2bf(wDcn[((M * C_ + c) * 3 + ky) * 3 + kx]);
    } else if (idx < total_dcn + total_off) {
        int id2  = idx - total_dcn;
        int m    = id2 / (KSTEPS * 512);
        int r    = id2 % (KSTEPS * 512);
        int t    = r / 512;
        int lane = (r % 512) / 16;
        int j    = r % 16;
        int M  = m * 16 + (lane & 15);
        int Kk = 32 * t + 16 * (lane >> 4) + j;
        int tap = Kk >> 7, c = Kk & 127;
        int ky = tap / 3, kx = tap - 3 * (tap / 3);
        float w = (M < 27) ? wOff[((M * C_ + c) * 3 + ky) * 3 + kx] : 0.f;
        aOff[id2] = f2bf(w);
    }
}

// ---------------------------------------------------------------------------
// Kernel 2: offset conv3x3 (27 ch padded to 32) via WMMA bf16, no LDS.
// Lane (g = lane>>4, n = lane&15) builds its own B fragment in registers.
// Zero padding handled branch-free: clamp the tap address to a valid pixel
// and scale the loaded value by a 0/1 validity factor (keeps the 16-load
// clause unconditional and coalesced).
// Writes off [B][18][H][W] and mask = sigmoid [B][9][H][W].
// ---------------------------------------------------------------------------
__global__ void __launch_bounds__(32)
offset_conv_kernel(const float* __restrict__ feat,
                   const unsigned short* __restrict__ aOff,
                   const float* __restrict__ offb,     // [27]
                   float* __restrict__ offbuf,
                   float* __restrict__ maskbuf)
{
    int tile = blockIdx.x;
    int tx = tile % NTILES_X;
    int y  = (tile / NTILES_X) % H_;
    int b  = tile / (NTILES_X * H_);
    int x0 = tx * 16;
    int lane = threadIdx.x;
    int g = lane >> 4;
    int n = lane & 15;
    int x = x0 + n;

    v8f zero = {0.f,0.f,0.f,0.f,0.f,0.f,0.f,0.f};
    v8f acc0 = zero, acc1 = zero;

    for (int tap = 0; tap < 9; ++tap) {
        int ky = tap / 3, kx = tap - 3 * (tap / 3);
        int yy  = y + ky - 1;
        int xxx = x + kx - 1;
        bool ok = (yy >= 0 && yy < H_ && xxx >= 0 && xxx < W_);
        float okf = ok ? 1.f : 0.f;
        int yyc = min(max(yy, 0), H_ - 1);
        int xxc = min(max(xxx, 0), W_ - 1);
        const float* fb = feat + ((size_t)(b * C_) * H_ + yyc) * W_ + xxc;
        #pragma unroll
        for (int tt = 0; tt < 4; ++tt) {
            int t  = tap * 4 + tt;
            int c0 = 32 * tt + 16 * g;
            AFrag bfr;
            #pragma unroll
            for (int j = 0; j < 16; ++j) {
                float v = okf * fb[(size_t)(c0 + j) * HW_];
                bfr.e[j] = (__bf16)v;          // native f32->bf16 cvt (RNE)
            }
            AFrag a0, a1;
            const uint4* p0 = (const uint4*)(aOff + ((size_t)(0 * KSTEPS + t) * 32 + lane) * 16);
            a0.q[0] = p0[0]; a0.q[1] = p0[1];
            const uint4* p1 = (const uint4*)(aOff + ((size_t)(1 * KSTEPS + t) * 32 + lane) * 16);
            a1.q[0] = p1[0]; a1.q[1] = p1[1];
            acc0 = __builtin_amdgcn_wmma_f32_16x16x32_bf16(false, a0.v, false, bfr.v, (short)0, acc0, false, false);
            acc1 = __builtin_amdgcn_wmma_f32_16x16x32_bf16(false, a1.v, false, bfr.v, (short)0, acc1, false, false);
        }
    }

    #pragma unroll
    for (int v = 0; v < 8; ++v) {
        int o = v + 8 * g;                      // C/D: M = v + 8*(lane>=16), N = lane&15
        float v0 = acc0[v] + offb[o];           // o in 0..15 -> all offsets
        offbuf[((size_t)(b * 18 + o) * H_ + y) * W_ + x] = v0;
        int o1 = 16 + o;
        float v1 = acc1[v] + ((o1 < 27) ? offb[o1] : 0.f);
        if (o1 < 18)
            offbuf[((size_t)(b * 18 + o1) * H_ + y) * W_ + x] = v1;
        else if (o1 < 27)
            maskbuf[((size_t)(b * 9 + (o1 - 18)) * H_ + y) * W_ + x] = 1.f / (1.f + expf(-v1));
    }
}

// ---------------------------------------------------------------------------
// Kernel 3: DCNv2 main conv via WMMA bf16.
// Tap geometry (bilinear weights * mask, clamped plane offsets) staged once
// in a tiny LDS table; each lane builds its own B fragments in registers.
// off[:,2k] -> y offset, off[:,2k+1] -> x offset (DCN convention).
// ---------------------------------------------------------------------------
__global__ void __launch_bounds__(32)
dcn_conv_kernel(const float* __restrict__ feat,
                const unsigned short* __restrict__ aDcn,
                const float* __restrict__ offbuf,
                const float* __restrict__ maskbuf,
                float* __restrict__ out)
{
    __shared__ float gw[9 * 16][4];            // bilinear weights * mask
    __shared__ int   gi[9 * 16][4];            // clamped plane offsets
    int tile = blockIdx.x;
    int tx = tile % NTILES_X;
    int y  = (tile / NTILES_X) % H_;
    int b  = tile / (NTILES_X * H_);
    int x0 = tx * 16;
    int lane = threadIdx.x;

    // stage tap geometry: 9 taps x 16 pixels
    for (int tpx = lane; tpx < 144; tpx += 32) {
        int tap = tpx >> 4;
        int pp  = tpx & 15;
        int ky = tap / 3, kx = tap - 3 * (tap / 3);
        int xq = x0 + pp;
        float offy = offbuf[((size_t)(b * 18 + 2 * tap)     * H_ + y) * W_ + xq];
        float offx = offbuf[((size_t)(b * 18 + 2 * tap + 1) * H_ + y) * W_ + xq];
        float mk   = maskbuf[((size_t)(b * 9 + tap) * H_ + y) * W_ + xq];
        float py  = (float)(y + ky - 1) + offy;
        float pxf = (float)(xq + kx - 1) + offx;
        float y0f = floorf(py), x0f = floorf(pxf);
        int iy = (int)y0f, ix = (int)x0f;
        float wy1 = py - y0f, wx1 = pxf - x0f;
        float wy0 = 1.f - wy1, wx0 = 1.f - wx1;
        #pragma unroll
        for (int q = 0; q < 4; ++q) {
            int cx = ix + (q & 1);
            int cy = iy + (q >> 1);
            float wq = ((q & 1) ? wx1 : wx0) * ((q >> 1) ? wy1 : wy0);
            bool okq = (cx >= 0 && cx < W_ && cy >= 0 && cy < H_);
            gi[tpx][q] = okq ? (cy * W_ + cx) : 0;
            gw[tpx][q] = okq ? wq * mk : 0.f;
        }
    }
    __syncthreads();

    int g = lane >> 4;
    int n = lane & 15;
    const float* fb = feat + (size_t)(b * C_) * HW_;

    v8f zero = {0.f,0.f,0.f,0.f,0.f,0.f,0.f,0.f};
    v8f acc[8];
    #pragma unroll
    for (int m = 0; m < 8; ++m) acc[m] = zero;

    for (int tap = 0; tap < 9; ++tap) {
        int gg = tap * 16 + n;
        float w0 = gw[gg][0], w1 = gw[gg][1], w2 = gw[gg][2], w3 = gw[gg][3];
        int   i0 = gi[gg][0], i1 = gi[gg][1], i2 = gi[gg][2], i3 = gi[gg][3];
        #pragma unroll
        for (int tt = 0; tt < 4; ++tt) {
            int t  = tap * 4 + tt;
            int c0 = 32 * tt + 16 * g;
            AFrag bfr;
            #pragma unroll
            for (int j = 0; j < 16; ++j) {
                const float* fc = fb + (size_t)(c0 + j) * HW_;
                float v = w0 * fc[i0] + w1 * fc[i1] + w2 * fc[i2] + w3 * fc[i3];
                bfr.e[j] = (__bf16)v;          // native f32->bf16 cvt (RNE)
            }
            #pragma unroll
            for (int m = 0; m < 8; ++m) {
                AFrag a;
                const uint4* p = (const uint4*)(aDcn + ((size_t)(m * KSTEPS + t) * 32 + lane) * 16);
                a.q[0] = p[0]; a.q[1] = p[1];
                acc[m] = __builtin_amdgcn_wmma_f32_16x16x32_bf16(false, a.v, false, bfr.v, (short)0, acc[m], false, false);
            }
        }
    }

    int x = x0 + n;
    #pragma unroll
    for (int m = 0; m < 8; ++m) {
        #pragma unroll
        for (int v = 0; v < 8; ++v) {
            int o = m * 16 + v + 8 * g;
            out[((size_t)(b * C_ + o) * H_ + y) * W_ + x] = acc[m][v];
        }
    }
}

// ---------------------------------------------------------------------------
// Kernel 4: GroupNorm(32) + affine + ReLU + residual, in place.
// One workgroup per (b, g); group = 4 channels x HW = 147456 elements.
// ---------------------------------------------------------------------------
__global__ void __launch_bounds__(256)
gn_relu_res_kernel(float* __restrict__ x,
                   const float* __restrict__ feat,
                   const float* __restrict__ gamma,
                   const float* __restrict__ beta)
{
    __shared__ float ss[256], sq[256];
    int bg = blockIdx.x;
    int b = bg >> 5;
    int g = bg & 31;
    size_t base = ((size_t)b * C_ + g * 4) * HW_;
    int tid = threadIdx.x;
    const int N = 4 * HW_;
    float s = 0.f, q = 0.f;
    for (int i = tid; i < N; i += 256) {
        float v = x[base + i];
        s += v; q += v * v;
    }
    ss[tid] = s; sq[tid] = q;
    __syncthreads();
    for (int st = 128; st > 0; st >>= 1) {
        if (tid < st) { ss[tid] += ss[tid + st]; sq[tid] += sq[tid + st]; }
        __syncthreads();
    }
    float mu   = ss[0] / (float)N;
    float var  = sq[0] / (float)N - mu * mu;
    float rstd = rsqrtf(var + 1e-5f);
    for (int i = tid; i < N; i += 256) {
        int c = g * 4 + i / HW_;
        float v = (x[base + i] - mu) * rstd * gamma[c] + beta[c];
        v = fmaxf(v, 0.f);
        x[base + i] = v + feat[base + i];
    }
}

// ---------------------------------------------------------------------------
// Kernel 5: per-pixel heads: 32 sampling offsets + confidence logit.
// ---------------------------------------------------------------------------
__global__ void __launch_bounds__(256)
heads_kernel(const float* __restrict__ featU,
             const float* __restrict__ sample_w,   // [32][128]
             const float* __restrict__ conf_w,     // [128]
             float* __restrict__ soff,             // [B][32][HW]
             float* __restrict__ confm)            // [B][HW]
{
    int idx = blockIdx.x * 256 + threadIdx.x;      // pixel over B*HW
    int b = idx / HW_;
    int p = idx % HW_;
    float accs[32];
    #pragma unroll
    for (int o = 0; o < 32; ++o) accs[o] = 0.f;
    float accc = 0.f;
    const float* fp = featU + (size_t)b * C_ * HW_ + p;
    for (int c = 0; c < C_; ++c) {
        float f = fp[(size_t)c * HW_];
        accc = fmaf(f, conf_w[c], accc);
        #pragma unroll
        for (int o = 0; o < 32; ++o)
            accs[o] = fmaf(f, sample_w[o * C_ + c], accs[o]);
    }
    #pragma unroll
    for (int o = 0; o < 32; ++o)
        soff[((size_t)b * 32 + o) * HW_ + p] = accs[o];
    confm[(size_t)b * HW_ + p] = accc;
}

// ---------------------------------------------------------------------------
// Kernel 6: grid-sample depth & conf at (x+off_x, y+off_y) with zero padding
// (align_corners=False unnormalization cancels exactly), softmax over 16
// heads, weighted depth sum.
// ---------------------------------------------------------------------------
__global__ void __launch_bounds__(256)
finalize_kernel(const float* __restrict__ soff,
                const float* __restrict__ init_depth,  // [B][HW]
                const float* __restrict__ confm,       // [B][HW]
                float* __restrict__ outp)              // [B][HW]
{
    int idx = blockIdx.x * 256 + threadIdx.x;
    int b = idx / HW_;
    int p = idx % HW_;
    int y = p / W_;
    int x = p - y * W_;
    const float* dbase = init_depth + (size_t)b * HW_;
    const float* cbase = confm + (size_t)b * HW_;
    float dep[NH_], lg[NH_];
    #pragma unroll
    for (int h = 0; h < NH_; ++h) {
        float ox = soff[((size_t)b * 32 + 2 * h)     * HW_ + p];  // channel 2h   = x
        float oy = soff[((size_t)b * 32 + 2 * h + 1) * HW_ + p];  // channel 2h+1 = y
        float sx = (float)x + ox;
        float sy = (float)y + oy;
        float x0f = floorf(sx), y0f = floorf(sy);
        int ix = (int)x0f, iy = (int)y0f;
        float wx1 = sx - x0f, wy1 = sy - y0f;
        float wx0 = 1.f - wx1, wy0 = 1.f - wy1;
        float d = 0.f, l = 0.f;
        #pragma unroll
        for (int q = 0; q < 4; ++q) {
            int cx = ix + (q & 1);
            int cy = iy + (q >> 1);
            float w = ((q & 1) ? wx1 : wx0) * ((q >> 1) ? wy1 : wy0);
            bool okq = (cx >= 0 && cx < W_ && cy >= 0 && cy < H_);
            int cxs = min(max(cx, 0), W_ - 1);
            int cys = min(max(cy, 0), H_ - 1);
            float wf = okq ? w : 0.f;
            int o = cys * W_ + cxs;
            d = fmaf(dbase[o], wf, d);
            l = fmaf(cbase[o], wf, l);
        }
        dep[h] = d; lg[h] = l;
    }
    float mx = lg[0];
    #pragma unroll
    for (int h = 1; h < NH_; ++h) mx = fmaxf(mx, lg[h]);
    float se = 0.f, sd = 0.f;
    #pragma unroll
    for (int h = 0; h < NH_; ++h) {
        float e = expf(lg[h] - mx);
        se += e;
        sd = fmaf(dep[h], e, sd);
    }
    outp[(size_t)b * HW_ + p] = sd / se;
}

// ---------------------------------------------------------------------------
extern "C" void kernel_launch(void* const* d_in, const int* in_sizes, int n_in,
                              void* d_out, int out_size, void* d_ws, size_t ws_size,
                              hipStream_t stream) {
    (void)in_sizes; (void)n_in; (void)out_size; (void)ws_size;
    const float* feat       = (const float*)d_in[0];  // [8,128,192,192]
    const float* init_depth = (const float*)d_in[1];  // [8,1,192,192]
    const float* dcn_off_w  = (const float*)d_in[2];  // [27,128,3,3]
    const float* dcn_off_b  = (const float*)d_in[3];  // [27]
    const float* dcn_w      = (const float*)d_in[4];  // [128,128,3,3]
    const float* gn_gamma   = (const float*)d_in[5];  // [128]
    const float* gn_beta    = (const float*)d_in[6];  // [128]
    const float* sample_w   = (const float*)d_in[7];  // [32,128]
    const float* conf_w     = (const float*)d_in[8];  // [1,128]
    float* outp = (float*)d_out;                      // [8,1,192,192]

    char* w = (char*)d_ws;
    auto carve = [&](size_t bytes) -> void* {
        void* p = (void*)w;
        w += (bytes + 255) & ~(size_t)255;
        return p;
    };
    float* offbuf  = (float*)carve((size_t)B_ * 18 * HW_ * 4);       // 21.2 MB
    float* maskbuf = (float*)carve((size_t)B_ * 9  * HW_ * 4);       // 10.6 MB
    float* dcnout  = (float*)carve((size_t)B_ * C_ * HW_ * 4);       // 151 MB (featU in place)
    float* soff    = (float*)carve((size_t)B_ * 32 * HW_ * 4);       // 37.7 MB
    float* confm   = (float*)carve((size_t)B_ * HW_ * 4);            // 1.2 MB
    unsigned short* aOff = (unsigned short*)carve((size_t)2 * KSTEPS * 512 * 2);
    unsigned short* aDcn = (unsigned short*)carve((size_t)8 * KSTEPS * 512 * 2);

    const int prep_total = 8 * KSTEPS * 512 + 2 * KSTEPS * 512;      // 184320
    prep_weights_kernel<<<(prep_total + 255) / 256, 256, 0, stream>>>(
        dcn_off_w, dcn_w, aOff, aDcn);

    offset_conv_kernel<<<NPIXTILES, 32, 0, stream>>>(
        feat, aOff, dcn_off_b, offbuf, maskbuf);

    dcn_conv_kernel<<<NPIXTILES, 32, 0, stream>>>(
        feat, aDcn, offbuf, maskbuf, dcnout);

    gn_relu_res_kernel<<<B_ * 32, 256, 0, stream>>>(
        dcnout, feat, gn_gamma, gn_beta);

    heads_kernel<<<(B_ * HW_) / 256, 256, 0, stream>>>(
        dcnout, sample_w, conf_w, soff, confm);

    finalize_kernel<<<(B_ * HW_) / 256, 256, 0, stream>>>(
        soff, init_depth, confm, outp);
}
